// DVR_85358180040849
// MI455X (gfx1250) — matrix operand order repaired and checked by reading, not compiled, not run on hardware
//
#include <hip/hip_runtime.h>

typedef _Float16 v16h __attribute__((ext_vector_type(16)));
typedef _Float16 v8h  __attribute__((ext_vector_type(8)));
typedef float    v8f  __attribute__((ext_vector_type(8)));

#define NSTEPS      16
#define NREFINE     8
#define FARV        2.4f
#define STEPV       (2.4f / 15.0f)
#define EPSF        1.1920929e-7f

// -------- dynamic LDS partition (bytes) --------
#define OFF_W2SW    0            // 32768 : W2 pre-swizzled into B-fragment order (f16)
#define OFF_W1R     32768        // 1536  : W1 rows 0..2  (f32 [3][128])
#define OFF_FB      34304        // 512   : b1 + feature @ W1[3:,:]  (per-batch constant)
#define OFF_B2      34816        // 512
#define OFF_W3      35328        // 512
#define OFF_WAVE    35840        // per wave: 64 B logit broadcast buffer
#define WAVE_BYTES  64
#define SMEM_BYTES  (OFF_WAVE + 4 * WAVE_BYTES)   // 36096 B

static __device__ __forceinline__ v16h ld_frag32(const _Float16* p0, const _Float16* p1) {
    v8h a = *(const v8h*)p0;
    v8h b = *(const v8h*)p1;
    v16h r;
#pragma unroll
    for (int i = 0; i < 8; ++i) { r[i] = a[i]; r[i + 8] = b[i]; }
    return r;
}

// Wave-cooperative MLP eval for a 16-point tile. Lane l owns ray m = l&15
// (both half-waves mirror the same rays). Returns the occupancy logit.
static __device__ __forceinline__ float mlp_eval(
    float px, float py, float pz, int lane,
    const float* __restrict__ w1r, const float* __restrict__ fb,
    const float* __restrict__ b2v, const float* __restrict__ w3v,
    const _Float16* __restrict__ w2sw,
    float* __restrict__ lg, float b3)
{
    const int m  = lane & 15;
    const int hi = lane >> 4;
    const int kb = 8 * hi;

    // ---- layer 1 (VALU) computed DIRECTLY into A-fragment registers ----
    // Lane l's A frags need row m, columns K = 16g + kb + {0..7}, g = 0..7
    // (16-bit A layout, ISA 7.12.2).  3 FMAs/col thanks to per-batch feature
    // folding; weight reads are half-wave-uniform LDS broadcasts.
    v16h A[4];
#pragma unroll
    for (int g = 0; g < 8; ++g) {
        const int base = 16 * g + kb;                      // 16B-aligned
        const float4 x0 = *(const float4*)(w1r + base);
        const float4 x1 = *(const float4*)(w1r + base + 4);
        const float4 y0 = *(const float4*)(w1r + 128 + base);
        const float4 y1 = *(const float4*)(w1r + 128 + base + 4);
        const float4 z0 = *(const float4*)(w1r + 256 + base);
        const float4 z1 = *(const float4*)(w1r + 256 + base + 4);
        const float4 f0 = *(const float4*)(fb + base);
        const float4 f1 = *(const float4*)(fb + base + 4);
        const int s = g >> 1;
        const int e = (g & 1) * 8;
        A[s][e + 0] = (_Float16)fmaxf(fmaf(px, x0.x, fmaf(py, y0.x, fmaf(pz, z0.x, f0.x))), 0.f);
        A[s][e + 1] = (_Float16)fmaxf(fmaf(px, x0.y, fmaf(py, y0.y, fmaf(pz, z0.y, f0.y))), 0.f);
        A[s][e + 2] = (_Float16)fmaxf(fmaf(px, x0.z, fmaf(py, y0.z, fmaf(pz, z0.z, f0.z))), 0.f);
        A[s][e + 3] = (_Float16)fmaxf(fmaf(px, x0.w, fmaf(py, y0.w, fmaf(pz, z0.w, f0.w))), 0.f);
        A[s][e + 4] = (_Float16)fmaxf(fmaf(px, x1.x, fmaf(py, y1.x, fmaf(pz, z1.x, f1.x))), 0.f);
        A[s][e + 5] = (_Float16)fmaxf(fmaf(px, x1.y, fmaf(py, y1.y, fmaf(pz, z1.y, f1.y))), 0.f);
        A[s][e + 6] = (_Float16)fmaxf(fmaf(px, x1.z, fmaf(py, y1.z, fmaf(pz, z1.z, f1.z))), 0.f);
        A[s][e + 7] = (_Float16)fmaxf(fmaf(px, x1.w, fmaf(py, y1.w, fmaf(pz, z1.w, f1.w))), 0.f);
    }

    // ---- layer 2 on WMMA (C starts from inline 0, b2 folded into epilogue),
    //      fused layer-3 partial dot on D frags ----
    float acc[8];
#pragma unroll
    for (int j = 0; j < 8; ++j) acc[j] = 0.f;
#pragma unroll
    for (int t = 0; t < 8; ++t) {
        float b2l = b2v[16 * t + m];
        float w3l = w3v[16 * t + m];
        v8f c = {};   // inline-0 C operand on first WMMA, accumulate in place after
#pragma unroll
        for (int s = 0; s < 4; ++s) {
            const _Float16* bp = w2sw + (size_t)((t * 4 + s) * 32 + lane) * 16;
            v16h B = ld_frag32(bp, bp + 8);
            c = __builtin_amdgcn_wmma_f32_16x16x32_f16(
                    false, A[s], false, B, (short)0, c, false, false);
        }
#pragma unroll
        for (int j = 0; j < 8; ++j) acc[j] += fmaxf(c[j] + b2l, 0.f) * w3l;
    }

    // ---- reduce over the 16 lanes of each half-wave ----
#pragma unroll
    for (int off = 1; off < 16; off <<= 1)
#pragma unroll
        for (int j = 0; j < 8; ++j) acc[j] += __shfl_xor(acc[j], off, 32);

    if (m == 0) {   // lane 0 -> rows 0..7, lane 16 -> rows 8..15
#pragma unroll
        for (int j = 0; j < 8; ++j) lg[hi * 8 + j] = acc[j] + b3;
    }
    asm volatile("s_wait_dscnt 0" ::: "memory");
    return lg[m];
}

__global__ void __launch_bounds__(128)
dvr_secant_kernel(const float* __restrict__ ray0, const float* __restrict__ dirs,
                  const float* __restrict__ feat, const float* __restrict__ W1,
                  const float* __restrict__ b1,   const float* __restrict__ W2,
                  const float* __restrict__ b2,   const float* __restrict__ W3,
                  const float* __restrict__ b3p,  float* __restrict__ out,
                  int N, int R)
{
    extern __shared__ char smem[];
    _Float16* w2sw = (_Float16*)(smem + OFF_W2SW);
    float*    w1r  = (float*)(smem + OFF_W1R);
    float*    fb   = (float*)(smem + OFF_FB);
    float*    b2v  = (float*)(smem + OFF_B2);
    float*    w3v  = (float*)(smem + OFF_W3);

    const int tid  = threadIdx.x;
    const int wave = tid >> 5;
    const int lane = tid & 31;
    float*    lg   = (float*)(smem + OFF_WAVE + wave * WAVE_BYTES);

    const int gBase = blockIdx.x * 64;       // 64 rays per block (4 waves x 16)
    const int bidx  = gBase / N;             // batch index (N % 64 == 0)

    // ---- cooperative setup ----
    // W2 -> B-fragment-swizzled LDS: frag (t,s), lane l holds halfs
// K = 32s + 8*(l/16) + (j&7) + 16*(j/8)  (j=0..15), column n = 16t + (l&15)
    for (int i = tid; i < 128 * 128; i += 128) {
        int j = i & 15, l = (i >> 4) & 31, s = (i >> 9) & 3, t = (i >> 11) & 7;
        int k = 32 * s + 8 * (l >> 4) + (j & 7) + 16 * (j >> 3);
        int n = 16 * t + (l & 15);
        w2sw[i] = (_Float16)W2[k * 128 + n];
    }
    for (int i = tid; i < 384; i += 128) w1r[i] = W1[i];   // rows 0..2 of W1
    {   // fb[n] = b1[n] + sum_f feat[b,f] * W1[3+f, n]
        float a = b1[tid];
#pragma unroll
        for (int f2 = 0; f2 < 32; ++f2)
            a = fmaf(feat[bidx * 32 + f2], W1[(3 + f2) * 128 + tid], a);
        fb[tid]  = a;
        b2v[tid] = b2[tid];
        w3v[tid] = W3[tid];
    }
    __syncthreads();
    const float b3 = b3p[0];

    // ---- per-wave ray tile ----
    const int m = lane & 15;
    const int g = gBase + wave * 16 + m;     // global ray id (both halves mirror)
    float ox = ray0[3 * g], oy = ray0[3 * g + 1], oz = ray0[3 * g + 2];
    float dx = dirs[3 * g], dy = dirs[3 * g + 1], dz = dirs[3 * g + 2];

    // ---- phase 1: march, online first outside->inside crossing ----
    float f0 = 0.f, f1 = 0.f, prevf = 0.f;
    float f_low = 0.f, f_high = 0.f, d_low = 0.f, d_high = 0.f;
    bool found = false;
#pragma unroll 1
    for (int s = 0; s < NSTEPS; ++s) {
        float d  = (float)s * STEPV;
        float fv = mlp_eval(ox + d * dx, oy + d * dy, oz + d * dz, lane,
                            w1r, fb, b2v, w3v, w2sw, lg, b3);
        if (s == 0) f0 = fv;
        if (s == 1) f1 = fv;
        bool cross = (s > 0) && !found && (prevf < 0.f) && (fv >= 0.f);
        if (cross) { found = true; f_low = prevf; f_high = fv; d_low = d - STEPV; d_high = d; }
        prevf = fv;
    }
    if (!found) { f_low = f0; f_high = f1; d_low = 0.f; d_high = STEPV; }  // idx==0 case

    // ---- phase 2: secant refinement ----
#pragma unroll 1
    for (int it = 0; it < NREFINE; ++it) {
        float dc  = f_high - f_low;
        float den = copysignf(fmaxf(fabsf(dc), EPSF), dc);
        float dn  = d_low - f_low * (d_high - d_low) / den;
        float fn  = mlp_eval(ox + dn * dx, oy + dn * dy, oz + dn * dz, lane,
                             w1r, fb, b2v, w3v, w2sw, lg, b3);
        bool below = fn < 0.f;
        d_low  = below ? dn : d_low;
        f_low  = below ? fn : f_low;
        d_high = below ? d_high : dn;
        f_high = below ? f_high : fn;
    }
    float dc  = f_high - f_low;
    float den = copysignf(fmaxf(fabsf(dc), EPSF), dc);
    float dp  = d_low - f_low * (d_high - d_low) / den;
    dp = found ? dp : FARV;

    if (lane < 16) {
        out[g]             = dp;                       // d_pred  [B,N]
        out[R + 3 * g + 0] = ox + dp * dx;             // p_pred  [B,N,3]
        out[R + 3 * g + 1] = oy + dp * dy;
        out[R + 3 * g + 2] = oz + dp * dz;
        out[4 * R + g]     = found ? 1.f : 0.f;        // mask    [B,N]
    }
}

extern "C" void kernel_launch(void* const* d_in, const int* in_sizes, int n_in,
                              void* d_out, int out_size, void* d_ws, size_t ws_size,
                              hipStream_t stream)
{
    const float* ray0 = (const float*)d_in[0];
    const float* dirs = (const float*)d_in[1];
    const float* feat = (const float*)d_in[2];
    const float* W1   = (const float*)d_in[3];
    const float* b1   = (const float*)d_in[4];
    const float* W2   = (const float*)d_in[5];
    const float* b2   = (const float*)d_in[6];
    const float* W3   = (const float*)d_in[7];
    const float* b3   = (const float*)d_in[8];
    float* out = (float*)d_out;

    const int R = in_sizes[0] / 3;        // total rays (B*N) = 65536
    const int B = in_sizes[2] / 32;       // batch
    const int N = R / B;                  // rays per batch = 16384

    const int blocks = R / 64;            // 64 rays / block (4 waves x 16)
    dvr_secant_kernel<<<blocks, 128, SMEM_BYTES, stream>>>(
        ray0, dirs, feat, W1, b1, W2, b2, W3, b3, out, N, R);
}
